// ZINCModel_81372450390514
// MI455X (gfx1250) — matrix operand order repaired and checked by reading, not compiled, not run on hardware
//
#include <hip/hip_runtime.h>

#define HD 128

typedef __attribute__((ext_vector_type(16))) __bf16 v16bf;
typedef __attribute__((ext_vector_type(8)))  float  v8f;

struct U8 { unsigned u[8]; };
struct WSrc { const float* p[18]; };

__device__ __forceinline__ unsigned short f2bf(float f) {
  unsigned u = __builtin_bit_cast(unsigned, f);
  u += 0x7FFFu + ((u >> 16) & 1u);
  return (unsigned short)(u >> 16);
}

__device__ __forceinline__ void fatomic(float* p, float v) {
  __hip_atomic_fetch_add(p, v, __ATOMIC_RELAXED, __HIP_MEMORY_SCOPE_AGENT);
}

__device__ __forceinline__ v8f vzero8() {
  v8f z = {0.f, 0.f, 0.f, 0.f, 0.f, 0.f, 0.f, 0.f};
  return z;
}

// low 32 bits of a generic pointer to LDS == LDS byte offset
__device__ __forceinline__ unsigned lds_off(const void* p) {
  return (unsigned)(uintptr_t)p;
}

// ---------------- elementwise / gather / scatter kernels ----------------

__global__ __launch_bounds__(256)
void node_embed_kernel(float* __restrict__ h0, const float* __restrict__ emb,
                       const int* __restrict__ x, int N) {
  int i = blockIdx.x * 256 + threadIdx.x;
  if (i >= N * HD) return;
  int n = i >> 7, h = i & 127;
  h0[i] = emb[x[n] * HD + h];
}

__global__ __launch_bounds__(256)
void edge_init_kernel(float* __restrict__ a, const float* __restrict__ h0,
                      const int* __restrict__ ei, const float* __restrict__ dist_emb,
                      int d, int E) {
  int i = blockIdx.x * 256 + threadIdx.x;
  if (i >= E * 32) return;
  int e = i >> 5, c = (i & 31) * 4;
  int u = ei[e], v = ei[E + e];
  const float4 de = *(const float4*)(dist_emb + d * HD + c);
  const float4 hu = *(const float4*)(h0 + (size_t)u * HD + c);
  const float4 hv = *(const float4*)(h0 + (size_t)v * HD + c);
  float4 r;
  r.x = de.x + hu.x + hv.x;
  r.y = de.y + hu.y + hv.y;
  r.z = de.z + hu.z + hv.z;
  r.w = de.w + hu.w + hv.w;
  *(float4*)(a + (size_t)e * HD + c) = r;
}

__global__ __launch_bounds__(256)
void edge_emb_kernel(float* __restrict__ ee, const float* __restrict__ tab,
                     const int* __restrict__ attr, int E) {
  int i = blockIdx.x * 256 + threadIdx.x;
  if (i >= E * HD) return;
  int e = i >> 7, h = i & 127;
  // edge_tables[h, attr[e], h] with layout (H,4,H)
  ee[i] = tab[h * (4 * HD) + attr[e] * HD + h];
}

__global__ __launch_bounds__(256)
void wconvert_kernel(unsigned short* __restrict__ dst, WSrc ws) {
  int m = blockIdx.y;
  int i = blockIdx.x * 256 + threadIdx.x;   // 0..16383
  int n = i >> 7, k = i & 127;
  // store transposed: dst[m][n][k] = W[k][n]
  dst[(size_t)m * 16384 + n * HD + k] = f2bf(ws.p[m][k * HD + n]);
}

__global__ __launch_bounds__(256)
void tri_scatter_kernel(float* __restrict__ S, const float* __restrict__ aI,
                        const float* __restrict__ aJ, const int* __restrict__ tri, int T) {
  int i = blockIdx.x * 256 + threadIdx.x;
  if (i >= T * 32) return;
  int t = i >> 5, c = (i & 31) * 4;
  int i0 = tri[t], i1 = tri[T + t], i2 = tri[2 * T + t];
  const float4 xa = *(const float4*)(aI + (size_t)i0 * HD + c);
  const float4 xb = *(const float4*)(aJ + (size_t)i1 * HD + c);
  float* o = S + (size_t)i2 * HD + c;
  fatomic(o + 0, xa.x * xb.x);
  fatomic(o + 1, xa.y * xb.y);
  fatomic(o + 2, xa.z * xb.z);
  fatomic(o + 3, xa.w * xb.w);
}

// agg[e] = (2h0[u])*(2h0[v]) + (ee?) + S[e] + S[inv[e]]
__global__ __launch_bounds__(256)
void agg_fin_kernel(float* __restrict__ agg, const float* __restrict__ h0,
                    const int* __restrict__ ei, const float* __restrict__ ee,
                    const float* __restrict__ S, const int* __restrict__ inv, int E) {
  int i = blockIdx.x * 256 + threadIdx.x;
  if (i >= E * 32) return;
  int e = i >> 5, c = (i & 31) * 4;
  int u = ei[e], v = ei[E + e], ie = inv[e];
  const float4 hu = *(const float4*)(h0 + (size_t)u * HD + c);
  const float4 hv = *(const float4*)(h0 + (size_t)v * HD + c);
  const float4 s0 = *(const float4*)(S + (size_t)e * HD + c);
  const float4 s1 = *(const float4*)(S + (size_t)ie * HD + c);
  float4 r;
  r.x = 4.f * hu.x * hv.x + s0.x + s1.x;
  r.y = 4.f * hu.y * hv.y + s0.y + s1.y;
  r.z = 4.f * hu.z * hv.z + s0.z + s1.z;
  r.w = 4.f * hu.w * hv.w + s0.w + s1.w;
  if (ee) {
    const float4 t = *(const float4*)(ee + (size_t)e * HD + c);
    r.x += t.x; r.y += t.y; r.z += t.z; r.w += t.w;
  }
  *(float4*)(agg + (size_t)e * HD + c) = r;
}

__global__ __launch_bounds__(256)
void add_inplace_kernel(float* __restrict__ dst, const float* __restrict__ src, int n4) {
  int i = blockIdx.x * 256 + threadIdx.x;
  if (i >= n4) return;
  float4 a = ((float4*)dst)[i];
  const float4 b = ((const float4*)src)[i];
  a.x += b.x; a.y += b.y; a.z += b.z; a.w += b.w;
  ((float4*)dst)[i] = a;
}

// ---------------- fused 128->128->128 MLP via bf16 WMMA ----------------
// Z[E,128] (f32) -> relu(Z@W1+b1) -> (.)@W2+b2 -> Y[E,128] (f32)
// W1T/W2T are bf16, stored transposed [n][k]; staged to LDS via async copies.

__device__ __forceinline__ v16bf loadA_frag(const unsigned short* lA, int aRow, int hi, int k0) {
  U8 f;
  const unsigned short* base = lA + aRow * HD + k0 + hi * 16;
  #pragma unroll
  for (int i = 0; i < 8; ++i) {
    int k = 2 * i;
    f.u[i] = *(const unsigned*)(base + k);
  }
  return __builtin_bit_cast(v16bf, f);
}

__device__ __forceinline__ v16bf loadB_frag(const unsigned short* lW, int n, int hi, int k0) {
  U8 f;
  const unsigned short* base = lW + n * HD + k0 + hi * 16;
  #pragma unroll
  for (int i = 0; i < 8; ++i) f.u[i] = *(const unsigned*)(base + 2 * i);
  return __builtin_bit_cast(v16bf, f);
}

__global__ __launch_bounds__(256)
void mlp_gemm_kernel(const float* __restrict__ Z,
                     const unsigned short* __restrict__ W1T, const float* __restrict__ b1,
                     const unsigned short* __restrict__ W2T, const float* __restrict__ b2,
                     float* __restrict__ Y, int E) {
  __shared__ __align__(16) unsigned short lA[64 * HD];
  __shared__ __align__(16) unsigned short lW1[HD * HD];
  __shared__ __align__(16) unsigned short lW2[HD * HD];

  const int tid = threadIdx.x;
  const int lane = tid & 31;
  const int wave = tid >> 5;
  const int row0 = blockIdx.x * 64;

  // stage weights (2x 128x128 bf16 = 64KB) via async global->LDS DMA (ASYNCcnt)
  for (int i = tid; i < 2048; i += 256) {
    unsigned goff = (unsigned)i * 16u;                // byte offset into weight matrix
    unsigned l1 = lds_off(&lW1[i * 8]);
    unsigned l2 = lds_off(&lW2[i * 8]);
    asm volatile("global_load_async_to_lds_b128 %0, %1, %2"
                 :: "v"(l1), "v"(goff), "s"(W1T) : "memory");
    asm volatile("global_load_async_to_lds_b128 %0, %1, %2"
                 :: "v"(l2), "v"(goff), "s"(W2T) : "memory");
  }

  // stage Z tile (64x128 f32) as bf16 (overlaps with async weight staging)
  for (int i = tid; i < 64 * 32; i += 256) {
    int r = i >> 5, c4 = i & 31;
    int row = row0 + r;
    float4 f = make_float4(0.f, 0.f, 0.f, 0.f);
    if (row < E) f = ((const float4*)(Z + (size_t)row * HD))[c4];
    unsigned short* p = &lA[r * HD + c4 * 4];
    p[0] = f2bf(f.x); p[1] = f2bf(f.y); p[2] = f2bf(f.z); p[3] = f2bf(f.w);
  }
  asm volatile("s_wait_asynccnt 0x0" ::: "memory");
  __syncthreads();

  // wave tiling: 2 row-halves (32 rows) x 4 col-quarters (32 cols) -> 2x2 16x16 tiles/wave
  const int rowHalf = wave & 1;
  const int colQ = wave >> 1;
  const int colbase = colQ * 32;
  const int n16 = lane & 15;
  const int hi = lane >> 4;
  const int aRow0 = rowHalf * 32 + n16;

  v8f acc[2][2];
  #pragma unroll
  for (int ri = 0; ri < 2; ++ri)
    #pragma unroll
    for (int ci = 0; ci < 2; ++ci) acc[ri][ci] = vzero8();

  // GEMM1: lA @ W1T
  #pragma unroll
  for (int kc = 0; kc < 4; ++kc) {
    const int k0 = kc * 32;
    v16bf a0 = loadA_frag(lA, aRow0, hi, k0);
    v16bf a1 = loadA_frag(lA, aRow0 + 16, hi, k0);
    v16bf b0 = loadB_frag(lW1, colbase + n16, hi, k0);
    v16bf b1f = loadB_frag(lW1, colbase + 16 + n16, hi, k0);
    acc[0][0] = __builtin_amdgcn_wmma_f32_16x16x32_bf16(false, a0, false, b0,  (short)0, acc[0][0], false, false);
    acc[0][1] = __builtin_amdgcn_wmma_f32_16x16x32_bf16(false, a0, false, b1f, (short)0, acc[0][1], false, false);
    acc[1][0] = __builtin_amdgcn_wmma_f32_16x16x32_bf16(false, a1, false, b0,  (short)0, acc[1][0], false, false);
    acc[1][1] = __builtin_amdgcn_wmma_f32_16x16x32_bf16(false, a1, false, b1f, (short)0, acc[1][1], false, false);
  }
  __syncthreads();
  // epilogue 1: relu(acc + b1) -> lA (bf16), becomes A of GEMM2
  #pragma unroll
  for (int ri = 0; ri < 2; ++ri) {
    #pragma unroll
    for (int ci = 0; ci < 2; ++ci) {
      const int n = colbase + ci * 16 + n16;
      const float bias = b1[n];
      #pragma unroll
      for (int i = 0; i < 8; ++i) {
        const int m = rowHalf * 32 + ri * 16 + i + hi * 8;
        float v = acc[ri][ci][i] + bias;
        v = v > 0.f ? v : 0.f;
        lA[m * HD + n] = f2bf(v);
      }
    }
  }
  __syncthreads();

  #pragma unroll
  for (int ri = 0; ri < 2; ++ri)
    #pragma unroll
    for (int ci = 0; ci < 2; ++ci) acc[ri][ci] = vzero8();

  // GEMM2: lA @ W2T
  #pragma unroll
  for (int kc = 0; kc < 4; ++kc) {
    const int k0 = kc * 32;
    v16bf a0 = loadA_frag(lA, aRow0, hi, k0);
    v16bf a1 = loadA_frag(lA, aRow0 + 16, hi, k0);
    v16bf b0 = loadB_frag(lW2, colbase + n16, hi, k0);
    v16bf b1f = loadB_frag(lW2, colbase + 16 + n16, hi, k0);
    acc[0][0] = __builtin_amdgcn_wmma_f32_16x16x32_bf16(false, a0, false, b0,  (short)0, acc[0][0], false, false);
    acc[0][1] = __builtin_amdgcn_wmma_f32_16x16x32_bf16(false, a0, false, b1f, (short)0, acc[0][1], false, false);
    acc[1][0] = __builtin_amdgcn_wmma_f32_16x16x32_bf16(false, a1, false, b0,  (short)0, acc[1][0], false, false);
    acc[1][1] = __builtin_amdgcn_wmma_f32_16x16x32_bf16(false, a1, false, b1f, (short)0, acc[1][1], false, false);
  }
  // epilogue 2: Y = acc + b2
  #pragma unroll
  for (int ri = 0; ri < 2; ++ri) {
    #pragma unroll
    for (int ci = 0; ci < 2; ++ci) {
      const int n = colbase + ci * 16 + n16;
      const float bias = b2[n];
      #pragma unroll
      for (int i = 0; i < 8; ++i) {
        const int m = rowHalf * 32 + ri * 16 + i + hi * 8;
        const int row = row0 + m;
        if (row < E) Y[(size_t)row * HD + n] = acc[ri][ci][i] + bias;
      }
    }
  }
}

// ---------------- batch-norm stats + normalize ----------------

__global__ __launch_bounds__(128)
void bn_stats_kernel(const float* __restrict__ Y, float* __restrict__ stats, int E) {
  const int h = threadIdx.x;
  const int r0 = blockIdx.x * 2048;
  int rend = r0 + 2048; if (rend > E) rend = E;
  float s = 0.f, q = 0.f;
  for (int r = r0; r < rend; ++r) {
    float y = Y[(size_t)r * HD + h];
    s += y; q += y * y;
  }
  fatomic(&stats[h], s);
  fatomic(&stats[HD + h], q);
}

__global__ __launch_bounds__(256)
void bn_relu_kernel(float* __restrict__ aout, const float* __restrict__ Y,
                    const float* __restrict__ stats, const float* __restrict__ g,
                    const float* __restrict__ beta, int E) {
  int i = blockIdx.x * 256 + threadIdx.x;
  if (i >= E * HD) return;
  int h = i & 127;
  const float invE = 1.f / (float)E;
  const float mean = stats[h] * invE;
  const float var = stats[HD + h] * invE - mean * mean;
  float v = g[h] * (Y[i] - mean) * rsqrtf(var + 1e-5f) + beta[h];
  aout[i] = v > 0.f ? v : 0.f;
}

// ---------------- pooling + readout ----------------

__global__ __launch_bounds__(256)
void count_kernel(float* __restrict__ counts, const int* __restrict__ batch0, int N) {
  int i = blockIdx.x * 256 + threadIdx.x;
  if (i >= N) return;
  fatomic(&counts[batch0[i]], 1.f);
}

__global__ __launch_bounds__(256)
void pool_nodes_kernel(float* __restrict__ pooled, const float* __restrict__ h0,
                       const int* __restrict__ batch0, int N) {
  int i = blockIdx.x * 256 + threadIdx.x;
  if (i >= N * 32) return;
  int n = i >> 5, c = (i & 31) * 4;
  int g = batch0[n];
  const float4 hv = *(const float4*)(h0 + (size_t)n * HD + c);
  float* o = pooled + (size_t)g * HD + c;
  fatomic(o + 0, 2.f * hv.x);
  fatomic(o + 1, 2.f * hv.y);
  fatomic(o + 2, 2.f * hv.z);
  fatomic(o + 3, 2.f * hv.w);
}

__global__ __launch_bounds__(256)
void pool_edges_kernel(float* __restrict__ pooled, const float* __restrict__ a,
                       const int* __restrict__ ei, const int* __restrict__ batch0, int E) {
  int i = blockIdx.x * 256 + threadIdx.x;
  if (i >= E * 32) return;
  int e = i >> 5, c = (i & 31) * 4;
  int g = batch0[ei[e]];
  const float4 av = *(const float4*)(a + (size_t)e * HD + c);
  float* o = pooled + (size_t)g * HD + c;
  fatomic(o + 0, av.x);
  fatomic(o + 1, av.y);
  fatomic(o + 2, av.z);
  fatomic(o + 3, av.w);
}

__global__ __launch_bounds__(64)
void final_mlp_kernel(float* __restrict__ out, const float* __restrict__ pooled,
                      const float* __restrict__ counts, const float* __restrict__ W1,
                      const float* __restrict__ b1, const float* __restrict__ W2,
                      const float* __restrict__ b2) {
  __shared__ float red[64];
  const int g = blockIdx.x;
  const int j = threadIdx.x;
  float cnt = counts[g];
  if (cnt < 1.f) cnt = 1.f;
  const float inv = 1.f / cnt;
  float acc = b1[j];
  const float* pr = pooled + (size_t)g * HD;
  for (int k = 0; k < HD; ++k) acc += pr[k] * inv * W1[k * 64 + j];
  const float h = acc > 0.f ? acc : (__expf(acc) - 1.f);  // elu
  red[j] = h * W2[j];
  __syncthreads();
  if (j == 0) {
    float s = 0.f;
    for (int k = 0; k < 64; ++k) s += red[k];
    out[g] = s + b2[0];
  }
}

// ---------------- host launcher ----------------

extern "C" void kernel_launch(void* const* d_in, const int* in_sizes, int n_in,
                              void* d_out, int out_size, void* d_ws, size_t ws_size,
                              hipStream_t stream) {
  const int N  = in_sizes[0];
  const int E1 = in_sizes[1] / 2;
  const int E2 = in_sizes[2] / 2;
  const int E3 = in_sizes[3] / 2;
  const int T  = in_sizes[9] / 3;
  const int G  = out_size;

  const int* x      = (const int*)d_in[0];
  const int* ei1    = (const int*)d_in[1];
  const int* ei2    = (const int*)d_in[2];
  const int* ei3    = (const int*)d_in[3];
  const int* eattr  = (const int*)d_in[4];
  const int* batch0 = (const int*)d_in[5];
  const int* inv1   = (const int*)d_in[6];
  const int* inv2   = (const int*)d_in[7];
  const int* t111   = (const int*)d_in[9];
  const int* t112   = (const int*)d_in[10];
  const int* t221   = (const int*)d_in[11];
  const int* t222   = (const int*)d_in[12];
  const int* t331   = (const int*)d_in[13];
  const float* node_emb = (const float*)d_in[14];
  const float* dist_emb = (const float*)d_in[15];
  const float* etab     = (const float*)d_in[16];
  const float* mlpW1 = (const float*)d_in[71];
  const float* mlpb1 = (const float*)d_in[72];
  const float* mlpW2 = (const float*)d_in[73];
  const float* mlpb2 = (const float*)d_in[74];
  float* out = (float*)d_out;

  auto LP = [&](int l, int d, int which) -> const float* {
    return (const float*)d_in[17 + (l * 3 + d) * 6 + which];
  };

  float* ws = (float*)d_ws;
  size_t off = 0;
  auto alloc = [&](size_t nf) { float* p = ws + off; off += nf; return p; };
  float* h0     = alloc((size_t)N * HD);
  float* a1     = alloc((size_t)E1 * HD);
  float* a2     = alloc((size_t)E2 * HD);
  float* a3     = alloc((size_t)E3 * HD);
  float* ee     = alloc((size_t)E1 * HD);
  float* agg1   = alloc((size_t)E1 * HD);
  float* agg2   = alloc((size_t)E2 * HD);
  float* Ybuf   = alloc((size_t)E3 * HD);   // doubles as triangle scatter buffer S
  float* pooled = alloc((size_t)G * HD);
  float* counts = alloc((size_t)G);
  float* stats  = alloc(256);
  unsigned short* wbf = (unsigned short*)(ws + off);   // 18 * 128*128 bf16 weights

  auto nb = [](long long n) { return (unsigned)((n + 255) / 256); };

  // init embeddings / edge features
  node_embed_kernel<<<nb((long long)N * HD), 256, 0, stream>>>(h0, node_emb, x, N);
  edge_init_kernel<<<nb((long long)E1 * 32), 256, 0, stream>>>(a1, h0, ei1, dist_emb, 0, E1);
  edge_init_kernel<<<nb((long long)E2 * 32), 256, 0, stream>>>(a2, h0, ei2, dist_emb, 1, E2);
  edge_init_kernel<<<nb((long long)E3 * 32), 256, 0, stream>>>(a3, h0, ei3, dist_emb, 2, E3);
  edge_emb_kernel<<<nb((long long)E1 * HD), 256, 0, stream>>>(ee, etab, eattr, E1);

  // convert all 18 layer weight matrices to transposed bf16
  WSrc wsrc;
  for (int l = 0; l < 3; ++l)
    for (int d = 0; d < 3; ++d) {
      wsrc.p[(l * 3 + d) * 2 + 0] = LP(l, d, 0);  // W1
      wsrc.p[(l * 3 + d) * 2 + 1] = LP(l, d, 2);  // W2
    }
  wconvert_kernel<<<dim3(64, 18), 256, 0, stream>>>(wbf, wsrc);

  auto run_mlp = [&](const float* Z, int l, int d, float* aout, int E) {
    const unsigned short* W1T = wbf + (size_t)((l * 3 + d) * 2 + 0) * 16384;
    const unsigned short* W2T = wbf + (size_t)((l * 3 + d) * 2 + 1) * 16384;
    hipMemsetAsync(stats, 0, 256 * sizeof(float), stream);
    mlp_gemm_kernel<<<(E + 63) / 64, 256, 0, stream>>>(Z, W1T, LP(l, d, 1), W2T, LP(l, d, 3),
                                                       Ybuf, E);
    bn_stats_kernel<<<(E + 2047) / 2048, 128, 0, stream>>>(Ybuf, stats, E);
    bn_relu_kernel<<<nb((long long)E * HD), 256, 0, stream>>>(aout, Ybuf, stats,
                                                              LP(l, d, 4), LP(l, d, 5), E);
  };

  for (int l = 0; l < 3; ++l) {
    // agg1: x0 product + edge_emb + symmetrized triangle scatters (targets E1)
    hipMemsetAsync(Ybuf, 0, (size_t)E1 * HD * sizeof(float), stream);
    tri_scatter_kernel<<<nb((long long)T * 32), 256, 0, stream>>>(Ybuf, a1, a1, t111, T);
    tri_scatter_kernel<<<nb((long long)T * 32), 256, 0, stream>>>(Ybuf, a2, a2, t221, T);
    tri_scatter_kernel<<<nb((long long)T * 32), 256, 0, stream>>>(Ybuf, a3, a3, t331, T);
    agg_fin_kernel<<<nb((long long)E1 * 32), 256, 0, stream>>>(agg1, h0, ei1, ee, Ybuf, inv1, E1);
    // agg2 (targets E2)
    hipMemsetAsync(Ybuf, 0, (size_t)E2 * HD * sizeof(float), stream);
    tri_scatter_kernel<<<nb((long long)T * 32), 256, 0, stream>>>(Ybuf, a1, a1, t112, T);
    tri_scatter_kernel<<<nb((long long)T * 32), 256, 0, stream>>>(Ybuf, a2, a2, t222, T);
    agg_fin_kernel<<<nb((long long)E2 * 32), 256, 0, stream>>>(agg2, h0, ei2, nullptr, Ybuf, inv2, E2);

    // updates (agg buffers captured old a's, so order below is safe)
    run_mlp(a3, l, 2, a3, E3);                                                 // d3: z = a3
    add_inplace_kernel<<<nb((long long)E1 * 32), 256, 0, stream>>>(agg1, a1, E1 * 32);
    run_mlp(agg1, l, 0, a1, E1);                                               // d1: z = a1+agg1
    add_inplace_kernel<<<nb((long long)E2 * 32), 256, 0, stream>>>(agg2, a2, E2 * 32);
    run_mlp(agg2, l, 1, a2, E2);                                               // d2: z = a2+agg2
  }

  // graph pooling + readout
  hipMemsetAsync(counts, 0, (size_t)G * sizeof(float), stream);
  hipMemsetAsync(pooled, 0, (size_t)G * HD * sizeof(float), stream);
  count_kernel<<<nb(N), 256, 0, stream>>>(counts, batch0, N);
  pool_nodes_kernel<<<nb((long long)N * 32), 256, 0, stream>>>(pooled, h0, batch0, N);
  pool_edges_kernel<<<nb((long long)E1 * 32), 256, 0, stream>>>(pooled, a1, ei1, batch0, E1);
  pool_edges_kernel<<<nb((long long)E2 * 32), 256, 0, stream>>>(pooled, a2, ei2, batch0, E2);
  pool_edges_kernel<<<nb((long long)E3 * 32), 256, 0, stream>>>(pooled, a3, ei3, batch0, E3);
  final_mlp_kernel<<<G, 64, 0, stream>>>(out, pooled, counts, mlpW1, mlpb1, mlpW2, mlpb2);

  (void)n_in; (void)ws_size; (void)inv1; (void)d_in;
}
